// GATModel_79319456023390
// MI455X (gfx1250) — compile-verified, matrix operand order checked by reading, not run on hardware
//
#include <hip/hip_runtime.h>
#include <stdint.h>

#define F_DIM   256
#define H_HEADS 8
#define C_CH    8
#define HC      64
#define NEG_SLOPE 0.2f
#define ENC_NEG_INF 0x007FFFFFu   // ordered-uint encoding of -inf

typedef __attribute__((ext_vector_type(16))) _Float16 v16h;
typedef __attribute__((ext_vector_type(8)))  float    v8f;
typedef unsigned int u32x4 __attribute__((ext_vector_type(4)));
typedef unsigned int u32x8 __attribute__((ext_vector_type(8)));

// Monotonic float<->uint mapping so atomicMax(u32) implements float max.
__device__ __forceinline__ unsigned enc_f(float f) {
  unsigned u = __float_as_uint(f);
  return (u & 0x80000000u) ? ~u : (u | 0x80000000u);
}
__device__ __forceinline__ float dec_f(unsigned u) {
  u = (u & 0x80000000u) ? (u & 0x7FFFFFFFu) : ~u;
  return __uint_as_float(u);
}

// ---------------------------------------------------------------------------
// Kernel 0: init accumulators (d_ws / d_out are poisoned by the harness).
// ---------------------------------------------------------------------------
__global__ void gat_init(float* __restrict__ out, float* __restrict__ ssum,
                         unsigned* __restrict__ mmax, int n_nodes) {
  int idx = blockIdx.x * blockDim.x + threadIdx.x;
  if (idx < n_nodes * HC) out[idx] = 0.0f;
  if (idx < n_nodes * H_HEADS) { ssum[idx] = 0.0f; mmax[idx] = ENC_NEG_INF; }
}

// ---------------------------------------------------------------------------
// Kernel 0b: decode int64 edge_index once into int32 src/dst (+ self loops).
// ---------------------------------------------------------------------------
__global__ void gat_edge_prep(const long long* __restrict__ ei, int E, int n_nodes,
                              int* __restrict__ src32, int* __restrict__ dst32) {
  int e = blockIdx.x * blockDim.x + threadIdx.x;
  if (e >= E + n_nodes) return;
  if (e < E) {
    src32[e] = (int)ei[e];
    dst32[e] = (int)ei[(size_t)E + e];
  } else {
    src32[e] = e - E;
    dst32[e] = e - E;
  }
}

// ---------------------------------------------------------------------------
// Kernel 1: h = x @ W^T via v_wmma_f32_16x16x32_f16.
// WG = 128 threads = 4 waves; each WG computes a 16-row x 64-col tile of h.
// Wave w owns output columns [16w, 16w+16). The 16x256 f32 x-tile is staged
// into LDS by ONE Tensor-Data-Mover descriptor (TENSORcnt path); the partial
// tail tile falls back to GLOBAL_LOAD_ASYNC_TO_LDS_B128 (ASYNCcnt path).
// The tile is converted to f16 in LDS, then consumed by WMMA.
// ---------------------------------------------------------------------------
__global__ __launch_bounds__(128) void gat_gemm_wmma(
    const float* __restrict__ x, const float* __restrict__ W,
    float* __restrict__ hbuf, int n_nodes) {
  // f32 tile (16KB) followed by f16 tile (8KB) in one LDS block.
  __shared__ __align__(16) unsigned char smem_raw[16 * F_DIM * 4 + 16 * F_DIM * 2];
  float*    tileF32 = reinterpret_cast<float*>(smem_raw);
  _Float16* tileF16 = reinterpret_cast<_Float16*>(smem_raw + 16 * F_DIM * 4);

  const int t    = threadIdx.x;
  const int lane = t & 31;
  const int wave = t >> 5;
  const int m    = lane & 15;   // row within 16x16 tile (A) / column (B,C)
  const int hi   = lane >> 4;   // half-wave selector

  // ---- Preload B fragments: column n = 16*wave + m, all K = 0..255. ----
  // 16-bit fragment K pattern per VGPR v (ISA 7.12.2):
  //   k = 32*kk + (v>=4 ? 16:0) + (v&3)*2 + hi*8, holding k and k+1.
  const int n = (wave << 4) + m;  // output channel 0..63
  v16h bfrag[8];
#pragma unroll
  for (int kk = 0; kk < 8; ++kk) {
#pragma unroll
    for (int v = 0; v < 8; ++v) {
      int k = kk * 32 + ((v & 4) << 2) + ((v & 3) << 1) + (hi << 3);
      float2 wv = *reinterpret_cast<const float2*>(&W[n * F_DIM + k]);
      bfrag[kk][2 * v]     = (_Float16)wv.x;
      bfrag[kk][2 * v + 1] = (_Float16)wv.y;
    }
  }

  const int  row0 = blockIdx.x * 16;
  const bool full = (row0 + 16 <= n_nodes);   // block-uniform

  // ---- Stage 1: bring 16x256 f32 tile into LDS. ----
  if (full) {
    if (wave == 0) {
      // Tensor DMA descriptor (D#), all fields wave-uniform.
      unsigned long long ga =
          (unsigned long long)(uintptr_t)(&x[(size_t)row0 * F_DIM]);
      unsigned lds_addr = (unsigned)(uintptr_t)tileF32;   // LDS byte offset
      unsigned tdim1    = (unsigned)(n_nodes - row0);     // rows available

      u32x4 g0;
      g0[0] = 1u;                                   // count=1, is_restore=0
      g0[1] = lds_addr;                             // lds_addr[31:0]
      g0[2] = (unsigned)ga;                         // global_addr[31:0]
      g0[3] = (unsigned)((ga >> 32) & 0x01FFFFFFull) | (2u << 30);  // type=2

      u32x8 g1;
      g1[0] = 2u << 16;                             // data_size=4B; mask=0
      g1[1] = ((unsigned)F_DIM & 0xFFFFu) << 16;    // tensor_dim0 lo16 (=256)
      g1[2] = (tdim1 & 0xFFFFu) << 16;              // tdim0 hi16=0 | tensor_dim1 lo16
      g1[3] = (tdim1 >> 16) | ((unsigned)F_DIM << 16);  // tdim1 hi16 | tile_dim0=256
      g1[4] = 16u;                                  // tile_dim1=16, tile_dim2=0
      g1[5] = (unsigned)F_DIM;                      // tensor_dim0_stride lo32 (=256)
      g1[6] = 0u;                                   // stride hi16=0, dim1_stride lo16=0
      g1[7] = 0u;                                   // dim1_stride hi32=0
      asm volatile("tensor_load_to_lds %0, %1" :: "s"(g0), "s"(g1) : "memory");
      __builtin_amdgcn_s_wait_tensorcnt(0);
    }
  } else {
    // Tail tile: per-lane async copies with clamped row addresses.
#pragma unroll
    for (int i = 0; i < 8; ++i) {
      int flat4 = i * 128 + t;          // 1024 x 16B covers the tile
      int r  = flat4 >> 6;
      int c4 = flat4 & 63;
      int rr = row0 + r; if (rr >= n_nodes) rr = n_nodes - 1;
      unsigned lds = (unsigned)(uintptr_t)(&tileF32[r * F_DIM + c4 * 4]);
      unsigned long long ga =
          (unsigned long long)(uintptr_t)(&x[(size_t)rr * F_DIM + c4 * 4]);
      asm volatile("global_load_async_to_lds_b128 %0, %1, off"
                   :: "v"(lds), "v"(ga) : "memory");
    }
    asm volatile("s_wait_asynccnt 0x0" ::: "memory");
  }
  __syncthreads();

  // ---- Stage 2: f32 -> f16 conversion inside LDS. ----
#pragma unroll
  for (int i = 0; i < 4; ++i) {
    int flat8 = i * 128 + t;            // 512 groups of 8 floats
    int base  = flat8 * 8;
    float4 a = *reinterpret_cast<const float4*>(&tileF32[base]);
    float4 b = *reinterpret_cast<const float4*>(&tileF32[base + 4]);
    _Float16* dp = &tileF16[base];
    dp[0] = (_Float16)a.x; dp[1] = (_Float16)a.y;
    dp[2] = (_Float16)a.z; dp[3] = (_Float16)a.w;
    dp[4] = (_Float16)b.x; dp[5] = (_Float16)b.y;
    dp[6] = (_Float16)b.z; dp[7] = (_Float16)b.w;
  }
  __syncthreads();

  // ---- 8 x WMMA over K = 256. ----
  v8f acc = {};
#pragma unroll
  for (int kk = 0; kk < 8; ++kk) {
    v16h afrag;
#pragma unroll
    for (int v = 0; v < 8; ++v) {
      int k = kk * 32 + ((v & 4) << 2) + ((v & 3) << 1) + (hi << 3);
      afrag[2 * v]     = tileF16[m * F_DIM + k];
      afrag[2 * v + 1] = tileF16[m * F_DIM + k + 1];
    }
    acc = __builtin_amdgcn_wmma_f32_16x16x32_f16(
        /*neg_a=*/false, afrag, /*neg_b=*/false, bfrag[kk],
        /*c_mod=*/(short)0, acc, /*reuse_a=*/false, /*reuse_b=*/false);
  }

  // ---- Store C: lane holds column n; VGPR v holds row hi*8 + v. ----
  if (full) {
    float* hp = &hbuf[((size_t)row0 + hi * 8) * HC + n];
#pragma unroll
    for (int v = 0; v < 8; ++v) hp[(size_t)v * HC] = acc[v];
  } else {
#pragma unroll
    for (int v = 0; v < 8; ++v) {
      int r = row0 + hi * 8 + v;
      if (r < n_nodes) hbuf[(size_t)r * HC + n] = acc[v];
    }
  }
}

// ---------------------------------------------------------------------------
// Kernel 2: per-node attention coefficients a_src/a_dst [N, H].
// ---------------------------------------------------------------------------
__global__ void gat_att_coeff(const float* __restrict__ hbuf,
                              const float* __restrict__ att_src,
                              const float* __restrict__ att_dst,
                              float* __restrict__ a_src, float* __restrict__ a_dst,
                              int n_nodes) {
  int idx = blockIdx.x * blockDim.x + threadIdx.x;
  if (idx >= n_nodes * H_HEADS) return;
  int node = idx >> 3, head = idx & 7;
  const float* hp = &hbuf[(size_t)node * HC + head * C_CH];
  float s = 0.0f, d = 0.0f;
#pragma unroll
  for (int c = 0; c < C_CH; ++c) {
    float v = hp[c];
    s += v * att_src[head * C_CH + c];
    d += v * att_dst[head * C_CH + c];
  }
  a_src[idx] = s;
  a_dst[idx] = d;
}

// ---------------------------------------------------------------------------
// Kernel 3: edge logits (LeakyReLU) + segment-max via ordered-uint atomicMax.
// ---------------------------------------------------------------------------
__global__ void gat_edge_logits(const int* __restrict__ src32,
                                const int* __restrict__ dst32,
                                const float* __restrict__ a_src,
                                const float* __restrict__ a_dst,
                                float* __restrict__ elog,
                                unsigned* __restrict__ mmax, int n_edges_tot) {
  int idx = blockIdx.x * blockDim.x + threadIdx.x;
  if (idx >= n_edges_tot * H_HEADS) return;
  int e = idx >> 3, head = idx & 7;
  int s = src32[e], d = dst32[e];
  float v = a_src[s * H_HEADS + head] + a_dst[d * H_HEADS + head];
  v = (v > 0.0f) ? v : v * NEG_SLOPE;
  elog[(size_t)e * H_HEADS + head] = v;
  atomicMax(&mmax[d * H_HEADS + head], enc_f(v));
}

// ---------------------------------------------------------------------------
// Kernel 4: e = exp(logit - max), segment-sum via f32 atomicAdd.
// ---------------------------------------------------------------------------
__global__ void gat_edge_expsum(const int* __restrict__ dst32,
                                float* __restrict__ elog,
                                const unsigned* __restrict__ mmax,
                                float* __restrict__ ssum, int n_edges_tot) {
  int idx = blockIdx.x * blockDim.x + threadIdx.x;
  if (idx >= n_edges_tot * H_HEADS) return;
  int e = idx >> 3, head = idx & 7;
  int d = dst32[e];
  float mv = dec_f(mmax[d * H_HEADS + head]);
  float p  = __expf(elog[(size_t)e * H_HEADS + head] - mv);
  elog[(size_t)e * H_HEADS + head] = p;
  atomicAdd(&ssum[d * H_HEADS + head], p);
}

// ---------------------------------------------------------------------------
// Kernel 5: weighted message scatter: out[dst] += alpha * h[src].
// 4 threads per edge; each handles 16 channels (2 heads) with float4 gathers.
// ---------------------------------------------------------------------------
__global__ void gat_aggregate(const int* __restrict__ src32,
                              const int* __restrict__ dst32,
                              const float* __restrict__ hbuf,
                              const float* __restrict__ elog,
                              const float* __restrict__ ssum,
                              float* __restrict__ out, int n_edges_tot) {
  int idx = blockIdx.x * blockDim.x + threadIdx.x;
  if (idx >= n_edges_tot * 4) return;
  int e = idx >> 2, q = idx & 3;
  int s = src32[e], d = dst32[e];
  int h0 = q * 2;
  float2 ev = *reinterpret_cast<const float2*>(&elog[(size_t)e * H_HEADS + h0]);
  float a0 = ev.x / (ssum[d * H_HEADS + h0]     + 1e-16f);
  float a1 = ev.y / (ssum[d * H_HEADS + h0 + 1] + 1e-16f);
  const float4* hp = reinterpret_cast<const float4*>(&hbuf[(size_t)s * HC + q * 16]);
  float* op = &out[(size_t)d * HC + q * 16];
#pragma unroll
  for (int j = 0; j < 4; ++j) {
    float4 hv = hp[j];
    float a = (j < 2) ? a0 : a1;
    atomicAdd(&op[4 * j + 0], hv.x * a);
    atomicAdd(&op[4 * j + 1], hv.y * a);
    atomicAdd(&op[4 * j + 2], hv.z * a);
    atomicAdd(&op[4 * j + 3], hv.w * a);
  }
}

// ---------------------------------------------------------------------------
// Kernel 6: in-place bias + log_softmax over the 64 channels of each node.
// ---------------------------------------------------------------------------
__global__ void gat_finalize(float* __restrict__ out, const float* __restrict__ bias,
                             int n_nodes) {
  int node = blockIdx.x * blockDim.x + threadIdx.x;
  if (node >= n_nodes) return;
  float* op = &out[(size_t)node * HC];
  float v[HC];
  float mx = -3.4e38f;
#pragma unroll
  for (int i = 0; i < HC; ++i) { v[i] = op[i] + bias[i]; mx = fmaxf(mx, v[i]); }
  float se = 0.0f;
#pragma unroll
  for (int i = 0; i < HC; ++i) se += __expf(v[i] - mx);
  float lse = mx + __logf(se);
#pragma unroll
  for (int i = 0; i < HC; ++i) op[i] = v[i] - lse;
}

// ---------------------------------------------------------------------------
extern "C" void kernel_launch(void* const* d_in, const int* in_sizes, int n_in,
                              void* d_out, int out_size, void* d_ws, size_t ws_size,
                              hipStream_t stream) {
  const float*     x       = (const float*)d_in[0];
  const long long* ei      = (const long long*)d_in[1];  // int64 [2, E]
  const float*     W       = (const float*)d_in[2];
  const float*     att_src = (const float*)d_in[3];
  const float*     att_dst = (const float*)d_in[4];
  const float*     bias    = (const float*)d_in[5];
  float*           out     = (float*)d_out;

  const int n_nodes = in_sizes[0] / F_DIM;   // 100000
  const int E       = in_sizes[1] / 2;       // 1600000
  const int ET      = E + n_nodes;           // with self-loops

  // Workspace carve-up (256B aligned).
  char* ws = (char*)d_ws;
  size_t off = 0;
  auto carve = [&](size_t bytes) -> void* {
    void* p = ws + off;
    off += (bytes + 255) & ~(size_t)255;
    return p;
  };
  float*    hbuf  = (float*)   carve((size_t)n_nodes * HC * sizeof(float));
  float*    a_src = (float*)   carve((size_t)n_nodes * H_HEADS * sizeof(float));
  float*    a_dst = (float*)   carve((size_t)n_nodes * H_HEADS * sizeof(float));
  unsigned* mmax  = (unsigned*)carve((size_t)n_nodes * H_HEADS * sizeof(unsigned));
  float*    ssum  = (float*)   carve((size_t)n_nodes * H_HEADS * sizeof(float));
  int*      src32 = (int*)     carve((size_t)ET * sizeof(int));
  int*      dst32 = (int*)     carve((size_t)ET * sizeof(int));
  float*    elog  = (float*)   carve((size_t)ET * H_HEADS * sizeof(float));
  (void)ws_size;

  const int BT = 256;

  // 0) init accumulators + decode edge index
  gat_init<<<(n_nodes * HC + BT - 1) / BT, BT, 0, stream>>>(out, ssum, mmax, n_nodes);
  gat_edge_prep<<<(ET + BT - 1) / BT, BT, 0, stream>>>(ei, E, n_nodes, src32, dst32);

  // 1) h = x @ W^T  (WMMA, TDM staging + async fallback)
  gat_gemm_wmma<<<(n_nodes + 15) / 16, 128, 0, stream>>>(x, W, hbuf, n_nodes);

  // 2) per-node attention coefficients
  gat_att_coeff<<<(n_nodes * H_HEADS + BT - 1) / BT, BT, 0, stream>>>(
      hbuf, att_src, att_dst, a_src, a_dst, n_nodes);

  // 3) edge logits + segment max
  gat_edge_logits<<<(ET * H_HEADS + BT - 1) / BT, BT, 0, stream>>>(
      src32, dst32, a_src, a_dst, elog, mmax, ET);

  // 4) exp + segment sum
  gat_edge_expsum<<<(ET * H_HEADS + BT - 1) / BT, BT, 0, stream>>>(
      dst32, elog, mmax, ssum, ET);

  // 5) weighted scatter-add of messages
  gat_aggregate<<<(ET * 4 + BT - 1) / BT, BT, 0, stream>>>(
      src32, dst32, hbuf, elog, ssum, out, ET);

  // 6) bias + log_softmax (in place on d_out)
  gat_finalize<<<(n_nodes + BT - 1) / BT, BT, 0, stream>>>(out, bias, n_nodes);
}
